// GlobalEmbeddingModelCore_87153476370858
// MI455X (gfx1250) — compile-verified, matrix operand order checked by reading, not run on hardware
//
#include <hip/hip_runtime.h>
#include <hip/hip_bf16.h>

// ---------------------------------------------------------------------------
// CDNA5 (gfx1250) implementation of GlobalEmbeddingModelCore.
// All dense GEMMs run on the matrix pipe via v_wmma_f32_16x16x32_f16.
// f16 activations feed WMMA; f32 copies feed the nonlinear gate math.
// Edge segment-sum is fused into the msg GEMM epilogue with atomicAdd(f32).
// Large-M GEMMs use a 32x64 tile/wave (8 WMMA per K-step, 2x W reuse);
// small sequential recurrent GEMMs use 16x64 for max wave parallelism.
// ---------------------------------------------------------------------------

typedef _Float16 half_t;
typedef __attribute__((ext_vector_type(16))) _Float16 v16h;
typedef __attribute__((ext_vector_type(8)))  float    v8f;

union Frag {
    v16h  v;
    uint4 q[2];
};

#define TPB 256

// ---------------------------------------------------------------------------
// WMMA GEMM:  C[M,N] = A[M,K] (f16, row-major) @ W[N,K]^T (f16, row-major)
//             + bias[N] (optional)
// If scat != null:  atomicAdd into C[scat[m]][n] instead of direct store
// (fused segment_sum for message passing).
// MT = number of 16-row A tiles per wave (1 or 2).  Each wave computes a
// (MT*16) x 64 output tile; 8 waves/block share the same 64-column W slab
// (L0-resident after the first wave).  M % (MT*16) == 0, N % 64 == 0,
// K % 32 == 0 required.
// ---------------------------------------------------------------------------
template <int MT>
__global__ __launch_bounds__(TPB) void k_gemm(
    const half_t* __restrict__ A,
    const half_t* __restrict__ W,
    const float*  __restrict__ bias,
    float*        __restrict__ C,
    const int*    __restrict__ scat,
    int M, int N, int K, int ldc)
{
    const int lane = threadIdx.x & 31;
    const int wave = threadIdx.x >> 5;
    const int mT   = blockIdx.x * 8 + wave;   // (MT*16)-row tile index (uniform)
    if (mT * (MT * 16) >= M) return;          // wave-uniform -> EXEC stays all-1
    const int nT   = blockIdx.y;              // 64-col tile index

    const int r  = lane & 15;
    const int kh = (lane >> 4) << 3;          // 0 for lanes 0-15, 8 for 16-31

    const half_t* Ar0 = A + (size_t)(mT * (MT * 16) + r) * K;
    const half_t* Ar1 = Ar0 + (size_t)16 * K;            // used when MT == 2
    const half_t* W0  = W + (size_t)(nT * 64 + r) * K;
    const size_t  ws  = (size_t)16 * K;                  // W row-tile stride

    v8f acc0 = {}, acc1 = {}, acc2 = {}, acc3 = {};
    v8f acc4 = {}, acc5 = {}, acc6 = {}, acc7 = {};

    for (int k = 0; k < K; k += 32) {
        Frag a0, a1, b0, b1, b2, b3;
        const half_t* ap0 = Ar0 + k + kh;
        const half_t* wp  = W0 + k + kh;
        a0.q[0] = *(const uint4*)(ap0);
        a0.q[1] = *(const uint4*)(ap0 + 16);
        if constexpr (MT == 2) {
            const half_t* ap1 = Ar1 + k + kh;
            a1.q[0] = *(const uint4*)(ap1);
            a1.q[1] = *(const uint4*)(ap1 + 16);
        }
        b0.q[0] = *(const uint4*)(wp);
        b0.q[1] = *(const uint4*)(wp + 16);
        b1.q[0] = *(const uint4*)(wp + ws);
        b1.q[1] = *(const uint4*)(wp + ws + 16);
        b2.q[0] = *(const uint4*)(wp + 2 * ws);
        b2.q[1] = *(const uint4*)(wp + 2 * ws + 16);
        b3.q[0] = *(const uint4*)(wp + 3 * ws);
        b3.q[1] = *(const uint4*)(wp + 3 * ws + 16);

        if (k + 32 < K) {   // speculative prefetch of next K-step
            __builtin_prefetch((const void*)(ap0 + 32), 0, 0);
            __builtin_prefetch((const void*)(wp + 32), 0, 0);
        }

        acc0 = __builtin_amdgcn_wmma_f32_16x16x32_f16(false, a0.v, false, b0.v,
                                                      (short)0, acc0, false, false);
        acc1 = __builtin_amdgcn_wmma_f32_16x16x32_f16(false, a0.v, false, b1.v,
                                                      (short)0, acc1, false, false);
        acc2 = __builtin_amdgcn_wmma_f32_16x16x32_f16(false, a0.v, false, b2.v,
                                                      (short)0, acc2, false, false);
        acc3 = __builtin_amdgcn_wmma_f32_16x16x32_f16(false, a0.v, false, b3.v,
                                                      (short)0, acc3, false, false);
        if constexpr (MT == 2) {
            acc4 = __builtin_amdgcn_wmma_f32_16x16x32_f16(false, a1.v, false, b0.v,
                                                          (short)0, acc4, false, false);
            acc5 = __builtin_amdgcn_wmma_f32_16x16x32_f16(false, a1.v, false, b1.v,
                                                          (short)0, acc5, false, false);
            acc6 = __builtin_amdgcn_wmma_f32_16x16x32_f16(false, a1.v, false, b2.v,
                                                          (short)0, acc6, false, false);
            acc7 = __builtin_amdgcn_wmma_f32_16x16x32_f16(false, a1.v, false, b3.v,
                                                          (short)0, acc7, false, false);
        }
    }

    // C/D layout: lane 0-15 -> col=lane, row=tilebase+v ; lane 16-31 -> row=+8
    const int rbase0 = mT * (MT * 16) + kh;
    const int rbase1 = rbase0 + 16;

#define STORE_TILE(ACC, J, RB)                                              \
    {                                                                       \
        const int ncol = nT * 64 + (J) * 16 + r;                            \
        const float bv = bias ? bias[ncol] : 0.0f;                          \
        _Pragma("unroll")                                                   \
        for (int v = 0; v < 8; ++v) {                                       \
            const float val = ACC[v] + bv;                                  \
            const int   row = (RB) + v;                                     \
            if (scat) atomicAdd(C + (size_t)scat[row] * ldc + ncol, val);   \
            else      C[(size_t)row * ldc + ncol] = val;                    \
        }                                                                   \
    }
    STORE_TILE(acc0, 0, rbase0)
    STORE_TILE(acc1, 1, rbase0)
    STORE_TILE(acc2, 2, rbase0)
    STORE_TILE(acc3, 3, rbase0)
    if constexpr (MT == 2) {
        STORE_TILE(acc4, 0, rbase1)
        STORE_TILE(acc5, 1, rbase1)
        STORE_TILE(acc6, 2, rbase1)
        STORE_TILE(acc7, 3, rbase1)
    }
#undef STORE_TILE
}

// ---------------------------------------------------------------------------
// Elementwise / data-movement helpers
// ---------------------------------------------------------------------------
__global__ void k_convert_f16(const float* __restrict__ in,
                              half_t* __restrict__ out, long n)
{
    long i = (long)blockIdx.x * blockDim.x + threadIdx.x;
    if (i < n) out[i] = (half_t)in[i];
}

// in: [R][Cc] f32 -> out: [Cc][R] f16
__global__ void k_transpose_f16(const float* __restrict__ in,
                                half_t* __restrict__ out, int R, int Cc)
{
    long i = (long)blockIdx.x * blockDim.x + threadIdx.x;
    if (i >= (long)R * Cc) return;
    int rr = (int)(i / Cc), cc = (int)(i % Cc);
    out[(size_t)cc * R + rr] = (half_t)in[i];
}

// out[row][:] = (f16) table[idx[row]][:]
__global__ void k_gather_f16(const float* __restrict__ table,
                             const int* __restrict__ idx,
                             half_t* __restrict__ out, long rows, int D_)
{
    long i = (long)blockIdx.x * blockDim.x + threadIdx.x;
    if (i >= rows * D_) return;
    long r = i / D_; int d = (int)(i % D_);
    out[i] = (half_t)table[(size_t)idx[r] * D_ + d];
}

// batch-major node init: node[g*T+t] = table[node_types[t*B+g]]  (f32 + f16)
__global__ void k_gather_node(const float* __restrict__ table,
                              const int* __restrict__ types,
                              float* __restrict__ n32, half_t* __restrict__ n16,
                              int B_, int T_, int D_)
{
    long i = (long)blockIdx.x * blockDim.x + threadIdx.x;
    long total = (long)B_ * T_ * D_;
    if (i >= total) return;
    long r = i / D_; int d = (int)(i % D_);
    int g = (int)(r / T_), t = (int)(r % T_);
    int packed = t * B_ + g;
    float val = table[(size_t)types[packed] * D_ + d];
    n32[i] = val;
    n16[i] = (half_t)val;
}

// concat buffer edge half (written once): cat[e][D + d] = e16[e][d]
__global__ void k_cat_edge(const half_t* __restrict__ e16,
                           half_t* __restrict__ cat, long E_, int D_)
{
    long i = (long)blockIdx.x * blockDim.x + threadIdx.x;
    if (i >= E_ * D_) return;
    long e = i / D_; int d = (int)(i % D_);
    cat[(size_t)e * (2 * D_) + D_ + d] = e16[i];
}

// concat buffer node half (per depth): cat[e][d] = n16[src[e]][d]
__global__ void k_cat_node(const half_t* __restrict__ n16,
                           const int* __restrict__ src,
                           half_t* __restrict__ cat, long E_, int D_)
{
    long i = (long)blockIdx.x * blockDim.x + threadIdx.x;
    if (i >= E_ * D_) return;
    long e = i / D_; int d = (int)(i % D_);
    cat[(size_t)e * (2 * D_) + d] = n16[(size_t)src[e] * D_ + d];
}

// shared GRU gate math (torch gate order r,z,n):
//   h' = (1-z)*tanh(xn + r*hn) + z*h
// writes f32 + f16 (and optional extra f16 copy = GRU layer output ys[t])
__global__ void k_gru_gate(const float* __restrict__ xp,
                           const float* __restrict__ hp,
                           float* __restrict__ h32,
                           half_t* __restrict__ h16,
                           half_t* __restrict__ extra16,
                           long M, int D_)
{
    long i = (long)blockIdx.x * blockDim.x + threadIdx.x;
    if (i >= M * D_) return;
    long m = i / D_; int d = (int)(i % D_);
    size_t b = (size_t)m * 3 * D_;
    float xr = xp[b + d], xz = xp[b + D_ + d], xn = xp[b + 2 * D_ + d];
    float hr = hp[b + d], hz = hp[b + D_ + d], hn = hp[b + 2 * D_ + d];
    float h  = h32[i];
    float rg = 1.0f / (1.0f + __expf(-(xr + hr)));
    float zg = 1.0f / (1.0f + __expf(-(xz + hz)));
    float ng = tanhf(xn + rg * hn);
    float out = (1.0f - zg) * ng + zg * h;
    h32[i] = out;
    h16[i] = (half_t)out;
    if (extra16) extra16[i] = (half_t)out;
}

// gated readout: cc16[g][512 + c] = sum_t sigmoid(G[g*32+t][c]) * P[g*32+t][c]
__global__ void k_readout(const float* __restrict__ G,
                          const float* __restrict__ P,
                          half_t* __restrict__ cc, int B_, int C2)
{
    long i = (long)blockIdx.x * blockDim.x + threadIdx.x;
    if (i >= (long)B_ * C2) return;
    int g = (int)(i / C2), c = (int)(i % C2);
    size_t base = (size_t)g * 32 * C2 + c;
    float s = 0.0f;
#pragma unroll 4
    for (int t = 0; t < 32; ++t) {
        float gv = G[base + (size_t)t * C2];
        float pv = P[base + (size_t)t * C2];
        s += pv / (1.0f + __expf(-gv));
    }
    cc[(size_t)g * 1536 + 512 + c] = (half_t)s;
}

// cc16[g][0..511] = global_emb f16
__global__ void k_copy_global(const half_t* __restrict__ src,
                              half_t* __restrict__ cc, int B_, int D_)
{
    long i = (long)blockIdx.x * blockDim.x + threadIdx.x;
    if (i >= (long)B_ * D_) return;
    int g = (int)(i / D_), d = (int)(i % D_);
    cc[(size_t)g * 1536 + d] = src[i];
}

// ---------------------------------------------------------------------------
// host side
// ---------------------------------------------------------------------------
static inline void launch_gemm(hipStream_t s, const half_t* A, const half_t* W,
                               const float* bias, float* C, const int* scat,
                               int M, int N, int K, int ldc)
{
    if (M >= 2048 && (M % 256) == 0) {
        // high-intensity 32x64 tile per wave (8 WMMA / K-step)
        dim3 grid(M / 256, N / 64);
        k_gemm<2><<<grid, TPB, 0, s>>>(A, W, bias, C, scat, M, N, K, ldc);
    } else {
        // latency-bound small GEMM: 16x64 tile, max wave parallelism
        dim3 grid((M + 127) / 128, N / 64);
        k_gemm<1><<<grid, TPB, 0, s>>>(A, W, bias, C, scat, M, N, K, ldc);
    }
}

static inline long gblk(long n) { return (n + TPB - 1) / TPB; }

extern "C" void kernel_launch(void* const* d_in, const int* in_sizes, int n_in,
                              void* d_out, int out_size, void* d_ws, size_t ws_size,
                              hipStream_t stream)
{
    const int Dm = 512, Tm = 32;
    const long Nn = in_sizes[18];          // 16384 nodes
    const long Ee = in_sizes[19];          // 32768 edges
    const int  Bb = (int)(Nn / Tm);        // 512 graphs

    // ---- inputs -----------------------------------------------------------
    const float* node_tab   = (const float*)d_in[0];
    const float* gru_w_ih   = (const float*)d_in[1];   // [3][1536][512]
    const float* gru_w_hh   = (const float*)d_in[2];
    const float* gru_b_ih   = (const float*)d_in[3];   // [3][1536]
    const float* gru_b_hh   = (const float*)d_in[4];
    const float* edge_tab   = (const float*)d_in[5];
    const float* msg_w      = (const float*)d_in[6];   // [1024][512]  (K,N)
    const float* msg_b      = (const float*)d_in[7];
    const float* cell_w_ih  = (const float*)d_in[8];   // [1536][512]  (N,K)
    const float* cell_w_hh  = (const float*)d_in[9];
    const float* cell_b_ih  = (const float*)d_in[10];
    const float* cell_b_hh  = (const float*)d_in[11];
    const float* gate_w     = (const float*)d_in[12];  // [512][1024]  (K,N)
    const float* gate_b     = (const float*)d_in[13];
    const float* proj_w     = (const float*)d_in[14];  // [512][1024]  (K,N)
    const float* proj_b     = (const float*)d_in[15];
    const float* merge_w    = (const float*)d_in[16];  // [1536][512]  (K,N)
    const float* merge_b    = (const float*)d_in[17];
    const int*   node_types = (const int*)d_in[18];
    const int*   edge_src   = (const int*)d_in[19];
    const int*   edge_dst   = (const int*)d_in[20];
    const int*   edge_types = (const int*)d_in[21];

    float* out_node   = (float*)d_out;                   // [N][512]
    float* out_merged = (float*)d_out + (size_t)Nn * Dm; // [B][512]

    // ---- workspace bump allocator ----------------------------------------
    uint8_t* base = (uint8_t*)d_ws;
    size_t off = 0;
    auto alloc = [&](size_t bytes) -> void* {
        void* p = base + off;
        off = (off + bytes + 255) & ~(size_t)255;
        return p;
    };

    half_t* w_ih16   = (half_t*)alloc((size_t)3 * 1536 * Dm * 2);
    half_t* w_hh16   = (half_t*)alloc((size_t)3 * 1536 * Dm * 2);
    half_t* cwih16   = (half_t*)alloc((size_t)1536 * Dm * 2);
    half_t* cwhh16   = (half_t*)alloc((size_t)1536 * Dm * 2);
    half_t* msgWT16  = (half_t*)alloc((size_t)Dm * (2 * Dm) * 2);   // [512][1024]
    half_t* gateWT16 = (half_t*)alloc((size_t)(2 * Dm) * Dm * 2);   // [1024][512]
    half_t* projWT16 = (half_t*)alloc((size_t)(2 * Dm) * Dm * 2);
    half_t* mergWT16 = (half_t*)alloc((size_t)Dm * 1536 * 2);       // [512][1536]

    half_t* pre16  = (half_t*)alloc((size_t)Nn * Dm * 2);   // time-major
    half_t* ysA16  = (half_t*)alloc((size_t)Nn * Dm * 2);
    half_t* ysB16  = (half_t*)alloc((size_t)Nn * Dm * 2);
    float*  h32    = (float*) alloc((size_t)Bb * Dm * 4);
    half_t* h16    = (half_t*)alloc((size_t)Bb * Dm * 2);
    float*  xp32   = (float*) alloc((size_t)Nn * 1536 * 4); // 96 MB, reused
    float*  hp32   = (float*) alloc((size_t)Nn * 1536 * 4); // 96 MB, reused
    half_t* node16 = (half_t*)alloc((size_t)Nn * Dm * 2);
    half_t* edge16 = (half_t*)alloc((size_t)Ee * Dm * 2);
    half_t* cat16  = (half_t*)alloc((size_t)Ee * 2 * Dm * 2);
    float*  agg32  = (float*) alloc((size_t)Nn * Dm * 4);
    half_t* agg16  = (half_t*)alloc((size_t)Nn * Dm * 2);
    half_t* cc16   = (half_t*)alloc((size_t)Bb * 1536 * 2);
    (void)ws_size;

    // ---- 0) weight conversion / transposition ----------------------------
    {
        long n;
        n = (long)3 * 1536 * Dm;
        k_convert_f16<<<gblk(n), TPB, 0, stream>>>(gru_w_ih, w_ih16, n);
        k_convert_f16<<<gblk(n), TPB, 0, stream>>>(gru_w_hh, w_hh16, n);
        n = (long)1536 * Dm;
        k_convert_f16<<<gblk(n), TPB, 0, stream>>>(cell_w_ih, cwih16, n);
        k_convert_f16<<<gblk(n), TPB, 0, stream>>>(cell_w_hh, cwhh16, n);
        n = (long)(2 * Dm) * Dm;
        k_transpose_f16<<<gblk(n), TPB, 0, stream>>>(msg_w,  msgWT16, 2 * Dm, Dm);
        k_transpose_f16<<<gblk(n), TPB, 0, stream>>>(gate_w, gateWT16, Dm, 2 * Dm);
        k_transpose_f16<<<gblk(n), TPB, 0, stream>>>(proj_w, projWT16, Dm, 2 * Dm);
        n = (long)1536 * Dm;
        k_transpose_f16<<<gblk(n), TPB, 0, stream>>>(merge_w, mergWT16, 1536, Dm);
    }

    // ---- 1) embeddings ----------------------------------------------------
    k_gather_f16<<<gblk(Nn * Dm), TPB, 0, stream>>>(node_tab, node_types, pre16, Nn, Dm);
    k_gather_node<<<gblk(Nn * Dm), TPB, 0, stream>>>(node_tab, node_types,
                                                     out_node, node16, Bb, Tm, Dm);
    k_gather_f16<<<gblk(Ee * Dm), TPB, 0, stream>>>(edge_tab, edge_types, edge16, Ee, Dm);

    // ---- 2) 3-layer GRU over time (global embedding) ----------------------
    const half_t* xin = pre16;
    half_t* ysbuf[2] = {ysA16, ysB16};
    for (int l = 0; l < 3; ++l) {
        // xp = X @ w_ih^T + b_ih  for all T*B rows at once
        launch_gemm(stream, xin, w_ih16 + (size_t)l * 1536 * Dm,
                    gru_b_ih + (size_t)l * 1536, xp32, nullptr,
                    (int)Nn, 1536, Dm, 1536);
        hipMemsetAsync(h32, 0, (size_t)Bb * Dm * 4, stream);
        hipMemsetAsync(h16, 0, (size_t)Bb * Dm * 2, stream);
        half_t* ycur = ysbuf[l & 1];
        for (int t = 0; t < Tm; ++t) {
            launch_gemm(stream, h16, w_hh16 + (size_t)l * 1536 * Dm,
                        gru_b_hh + (size_t)l * 1536, hp32, nullptr,
                        Bb, 1536, Dm, 1536);
            k_gru_gate<<<gblk((long)Bb * Dm), TPB, 0, stream>>>(
                xp32 + (size_t)t * Bb * 1536, hp32, h32, h16,
                ycur + (size_t)t * Bb * Dm, Bb, Dm);
        }
        xin = ycur;
    }
    const half_t* global16 = xin + (size_t)(Tm - 1) * Bb * Dm;  // [B][512] f16

    // ---- 3) message passing (DEPTH=3) -------------------------------------
    k_cat_edge<<<gblk(Ee * Dm), TPB, 0, stream>>>(edge16, cat16, Ee, Dm);
    for (int depth = 0; depth < 3; ++depth) {
        k_cat_node<<<gblk(Ee * Dm), TPB, 0, stream>>>(node16, edge_src, cat16, Ee, Dm);
        hipMemsetAsync(agg32, 0, (size_t)Nn * Dm * 4, stream);
        // agg[dst] += concat(node[src], edge_emb) @ msg_w + msg_b   (fused)
        launch_gemm(stream, cat16, msgWT16, msg_b, agg32, edge_dst,
                    (int)Ee, Dm, 2 * Dm, Dm);
        k_convert_f16<<<gblk(Nn * Dm), TPB, 0, stream>>>(agg32, agg16, Nn * Dm);
        // GRUCell(agg, node)
        launch_gemm(stream, agg16, cwih16, cell_b_ih, xp32, nullptr,
                    (int)Nn, 1536, Dm, 1536);
        launch_gemm(stream, node16, cwhh16, cell_b_hh, hp32, nullptr,
                    (int)Nn, 1536, Dm, 1536);
        k_gru_gate<<<gblk(Nn * Dm), TPB, 0, stream>>>(
            xp32, hp32, out_node, node16, nullptr, Nn, Dm);
    }

    // ---- 4) gated graph readout + merge -----------------------------------
    launch_gemm(stream, node16, gateWT16, gate_b, hp32, nullptr,
                (int)Nn, 2 * Dm, Dm, 2 * Dm);                 // pre-sigmoid gate
    launch_gemm(stream, node16, projWT16, proj_b, xp32, nullptr,
                (int)Nn, 2 * Dm, Dm, 2 * Dm);                 // projection
    k_readout<<<gblk((long)Bb * 2 * Dm), TPB, 0, stream>>>(hp32, xp32, cc16, Bb, 2 * Dm);
    k_copy_global<<<gblk((long)Bb * Dm), TPB, 0, stream>>>(global16, cc16, Bb, Dm);
    // merged = concat(global_emb, graph_emb) @ merge_w + merge_b
    launch_gemm(stream, cc16, mergWT16, merge_b, out_merged, nullptr,
                Bb, Dm, 1536, Dm);
}
